// MultiViewGCN_43473658970438
// MI455X (gfx1250) — matrix operand order repaired and checked by reading, not compiled, not run on hardware
//
#include <hip/hip_runtime.h>
#include <stdint.h>

#define NN 50000
#define NE 800000
#define DD 128

typedef __attribute__((ext_vector_type(16))) _Float16 v16h;
typedef __attribute__((ext_vector_type(8)))  float    v8f;

union HFrag { uint4 q[2]; v16h v; };
union FAcc  { float f[8]; v8f  v; };

// ---------------- degree / norm ----------------
__global__ void deg_kernel(const int* __restrict__ dst, float* __restrict__ deg, int n_edges) {
    int e = blockIdx.x * blockDim.x + threadIdx.x;
    if (e < n_edges) atomicAdd(&deg[dst[e]], 1.0f);
}

__global__ void norm_kernel(const float* __restrict__ deg, float* __restrict__ nrm, int n) {
    int i = blockIdx.x * blockDim.x + threadIdx.x;
    if (i < n) nrm[i] = rsqrtf(deg[i]);   // deg^-0.5 ; deg==0 -> inf (matches reference)
}

// ---------------- edge scatter: agg[dst] += feat[src] * norm[src] ----------------
// one wave (32 lanes) per edge; each lane moves a float4 (128 floats per edge)
__global__ void scatter_kernel(const float* __restrict__ feat, const float* __restrict__ nrm,
                               const int* __restrict__ src, const int* __restrict__ dst,
                               float* __restrict__ agg, int n_edges) {
    int lane = threadIdx.x & 31;
    int e = blockIdx.x * (blockDim.x >> 5) + (threadIdx.x >> 5);
    if (e >= n_edges) return;
    int s = src[e];
    int d = dst[e];
    float ns = nrm[s];
    const float4* f4 = (const float4*)(feat + (size_t)s * DD);
    float4 v = f4[lane];
    float* a = agg + (size_t)d * DD + lane * 4;
    atomicAdd(a + 0, v.x * ns);
    atomicAdd(a + 1, v.y * ns);
    atomicAdd(a + 2, v.z * ns);
    atomicAdd(a + 3, v.w * ns);
}

// ---------------- pre-swizzle W (f32 [128,128], out[j,k]) into WMMA B fragments ----
// B tile (32x16, f16, wave32): lane L holds N = nt*16 + (L&15); half h holds K = kt*32 + (h<8 ? 8*(L>>4)+h : 16+8*(L>>4)+h-8).
// Layout: Wpre[((nt*4 + kt)*32 + lane)*16 + h] = (f16) W[(nt*16+(lane&15))*128 + K]
__global__ void wprep_kernel(const float* __restrict__ W, _Float16* __restrict__ Wpre) {
    int idx  = blockIdx.x * blockDim.x + threadIdx.x;  // 0..1023
    int lane = idx & 31;
    int frag = idx >> 5;          // nt*4 + kt
    int nt = frag >> 2;
    int kt = frag & 3;
    int j  = nt * 16 + (lane & 15);
    _Float16* o = Wpre + (size_t)idx * 16;
#pragma unroll
    for (int h = 0; h < 16; ++h) {
        int k = kt * 32 + (h < 8 ? 8 * (lane >> 4) + h : 16 + 8 * (lane >> 4) + (h - 8));
        o[h] = (_Float16)W[j * DD + k];
    }
}

// ---------------- dense projection: out = relu((agg*norm) @ W^T + b) -------------
// block = 256 thr = 8 waves; block covers 16 node-rows x 128 output cols.
// Wave w computes the 16x16 tile at columns [16w,16w+16) with 4 chained
// v_wmma_f32_16x16x32_f16 ops over K=128.
__global__ void gemm_kernel(const float* __restrict__ agg, const float* __restrict__ nrm,
                            const _Float16* __restrict__ Wpre, const float* __restrict__ bias,
                            float* __restrict__ out) {
    __shared__ __align__(16) _Float16 sA[16 * DD];   // 16 rows x 128 K, f16 (4 KB)
    int tid  = threadIdx.x;
    int base = blockIdx.x * 16;

    // stage 16x128 f32 -> f16 into LDS, fusing the norm[dst] post-scale
#pragma unroll
    for (int i = 0; i < 8; ++i) {
        int idx = i * 256 + tid;           // 0..2047
        int row = idx >> 7;
        int col = idx & 127;
        float v = agg[(size_t)(base + row) * DD + col] * nrm[base + row];
        sA[row * DD + col] = (_Float16)v;
    }
    __syncthreads();

    int lane = tid & 31;
    int nt   = tid >> 5;        // wave id == output column tile
    int row  = lane & 15;       // A-matrix M row for this lane
    int hi   = lane >> 4;       // lane group

    FAcc acc;
#pragma unroll
    for (int r = 0; r < 8; ++r) acc.f[r] = 0.0f;

#pragma unroll
    for (int kt = 0; kt < 4; ++kt) {
        // A fragment (16x32 f16): lanes 0-15 hold K = kt*32 + {0..7, 16..23},
        //                         lanes 16-31 hold K = kt*32 + {8..15, 24..31}
        int ka = kt * 32 + hi * 8;
        const uint4* pa = (const uint4*)(sA + row * DD + ka);  // 16B aligned
        HFrag a;
        a.q[0] = pa[0];     // halves ka .. ka+7
        a.q[1] = pa[2];     // halves ka+16 .. ka+23
        // B fragment: two 16B loads from the pre-swizzled weights
        const uint4* pb = (const uint4*)(Wpre + (size_t)((nt * 4 + kt) * 32 + lane) * 16);
        HFrag b;
        b.q[0] = pb[0];
        b.q[1] = pb[1];
        acc.v = __builtin_amdgcn_wmma_f32_16x16x32_f16(
            /*neg_a=*/false, a.v, /*neg_b=*/false, b.v,
            /*c_mod=*/(short)0, acc.v, /*reuse_a=*/false, /*reuse_b=*/false);
    }

    // C/D layout: VGPR r, lane L -> M = r + 8*(L>=16), N = L&15
    int j  = nt * 16 + (lane & 15);
    float bj = bias[j];
#pragma unroll
    for (int r = 0; r < 8; ++r) {
        int node = base + r + 8 * hi;
        float v  = acc.f[r] + bj;
        out[(size_t)node * DD + j] = v > 0.0f ? v : 0.0f;
    }
}

// ---------------- launcher ----------------
extern "C" void kernel_launch(void* const* d_in, const int* in_sizes, int n_in,
                              void* d_out, int out_size, void* d_ws, size_t ws_size,
                              hipStream_t stream) {
    (void)in_sizes; (void)n_in; (void)out_size; (void)ws_size;
    char* ws = (char*)d_ws;

    constexpr size_t SZ_N  = (size_t)NN * 4;        // 200,000 B
    constexpr size_t SZ_ND = (size_t)NN * DD * 4;   // 25,600,000 B
    constexpr size_t AL_N  = (SZ_N + 255) & ~(size_t)255;

    float*     deg  = (float*)(ws);
    float*     nrm  = (float*)(ws + AL_N);
    float*     agg  = (float*)(ws + 2 * AL_N);
    float*     h1   = (float*)(ws + 2 * AL_N + SZ_ND);
    _Float16*  Wpre = (_Float16*)(ws + 2 * AL_N + 2 * SZ_ND);
    // total workspace use: ~51.7 MB

    for (int v = 0; v < 3; ++v) {
        const float* feat = (const float*)d_in[7 * v + 0];
        const float* W1   = (const float*)d_in[7 * v + 1];
        const float* b1   = (const float*)d_in[7 * v + 2];
        const float* W2   = (const float*)d_in[7 * v + 3];
        const float* b2   = (const float*)d_in[7 * v + 4];
        const int*   src  = (const int*)d_in[7 * v + 5];
        const int*   dst  = (const int*)d_in[7 * v + 6];
        float* outv = (float*)d_out + (size_t)v * NN * DD;

        // degree + normalization (shared by both layers of this view)
        hipMemsetAsync(deg, 0, SZ_N, stream);
        deg_kernel<<<(NE + 255) / 256, 256, 0, stream>>>(dst, deg, NE);
        norm_kernel<<<(NN + 255) / 256, 256, 0, stream>>>(deg, nrm, NN);

        // layer 1
        hipMemsetAsync(agg, 0, SZ_ND, stream);
        scatter_kernel<<<NE / 8, 256, 0, stream>>>(feat, nrm, src, dst, agg, NE);
        wprep_kernel<<<4, 256, 0, stream>>>(W1, Wpre);
        gemm_kernel<<<NN / 16, 256, 0, stream>>>(agg, nrm, Wpre, b1, h1);

        // layer 2
        hipMemsetAsync(agg, 0, SZ_ND, stream);
        scatter_kernel<<<NE / 8, 256, 0, stream>>>(h1, nrm, src, dst, agg, NE);
        wprep_kernel<<<4, 256, 0, stream>>>(W2, Wpre);
        gemm_kernel<<<NN / 16, 256, 0, stream>>>(agg, nrm, Wpre, b2, outv);
    }
}
